// GPT_25254407700592
// MI455X (gfx1250) — compile-verified, matrix operand order checked by reading, not run on hardware
//
#include <hip/hip_runtime.h>

// ---------------------------------------------------------------------------
// Mini-transformer forward for MI455X (gfx1250).
//  * All GEMMs + attention matmuls on v_wmma_f32_16x16x32_bf16 (wave32).
//  * GEMM tiles DMA'd straight into LDS with global_load_async_to_lds_b128
//    (ASYNCcnt-tracked, double buffered, 64-K staging -> 16 WMMA per barrier).
//  * Weights pre-converted to bf16 once per launch; fp32 masters and bf16
//    copies both fit in the 192MB L2, so the GEMMs are WMMA-issue bound.
// ---------------------------------------------------------------------------

typedef __attribute__((ext_vector_type(16))) __bf16 v16bf;
typedef __attribute__((ext_vector_type(4)))  __bf16 v4bf;
typedef __attribute__((ext_vector_type(8)))  float  v8f;

#define D_MODEL 512
#define NHEAD   8
#define DHEAD   64
#define NLAYER  11
#define HIDDEN  1536
#define BATCH   8
#define SEQ     1024
#define ROWS    (BATCH * SEQ)   // 8192
#define VOCAB   1024
#define NBIG    2048
#define DBIG    128

__device__ __forceinline__ v8f wmma_bf16(v16bf a, v16bf b, v8f c) {
  return __builtin_amdgcn_wmma_f32_16x16x32_bf16(
      /*neg_a=*/false, a, /*neg_b=*/false, b,
      /*c_mod=*/(short)0, c, /*reuse_a=*/false, /*reuse_b=*/false);
}

// Async DMA: 16 bytes global -> LDS, GVS addressing (SGPR base + VGPR offset).
__device__ __forceinline__ void async_ld_b128(void* lds_dst, unsigned gvoff,
                                              const void* gbase) {
  unsigned ldsa =
      (unsigned)(unsigned long long)(__attribute__((address_space(3))) void*)lds_dst;
  asm volatile("global_load_async_to_lds_b128 %0, %1, %2"
               :
               : "v"(ldsa), "v"(gvoff),
                 "s"((unsigned long long)(size_t)gbase)
               : "memory");
}
#define ASYNC_WAIT_8() asm volatile("s_wait_asynccnt 0x8" ::: "memory")
#define ASYNC_WAIT_0() asm volatile("s_wait_asynccnt 0x0" ::: "memory")

// ---------------------------------------------------------------------------
// GEMM: C[M,N] = A[M,K](bf16) @ W[N,K](bf16)^T, fused epilogue by template:
//   MODE 0: C = A W^T                 (f32)
//   MODE 1: C = R + A W^T             (f32, residual add)
//   MODE 2: Cbf = (relu(A W^T))^2     (bf16)
// 256 thr = 8 waves (4m x 2n); block tile 128x128; wave tile 32x64.
// LDS stage = 64 K-columns, double buffered: 16 WMMA per barrier pair.
// Per thread per stage: 4 async b128 for A + 4 for W -> wait threshold 8.
// ---------------------------------------------------------------------------
template <int MODE>
__global__ __launch_bounds__(256)
void gemm_bf16_k(const __bf16* __restrict__ A, const __bf16* __restrict__ W,
                 float* __restrict__ C, const float* __restrict__ R,
                 __bf16* __restrict__ Cbf, int N, int K) {
  __shared__ __bf16 As[2][128 * 64];   // 16KB per buffer
  __shared__ __bf16 Bs[2][128 * 64];

  const int tid   = threadIdx.x;
  const int m_blk = blockIdx.y * 128;
  const int n_blk = blockIdx.x * 128;
  const int wid   = tid >> 5;
  const int lane  = tid & 31;
  const int lcol  = lane & 15;
  const int lhalf = lane >> 4;
  const int wm    = (wid >> 1) * 32;   // 0,32,64,96
  const int wn    = (wid & 1) * 64;    // 0,64

  // DMA assignment: 2 threads per row, 32 contiguous bf16 (64B = 4 x b128)
  const int ld_row  = tid >> 1;        // 0..127
  const int ld_half = tid & 1;
  const unsigned a_off0 = (unsigned)((m_blk + ld_row) * K + ld_half * 32) * 2u;
  const unsigned w_off0 = (unsigned)((n_blk + ld_row) * K + ld_half * 32) * 2u;
  __bf16* a_dst = &As[0][ld_row * 64 + ld_half * 32];
  __bf16* w_dst = &Bs[0][ld_row * 64 + ld_half * 32];

  auto issue = [&](int slot, int k0) {
    const unsigned kb = (unsigned)k0 * 2u;
#pragma unroll
    for (int j = 0; j < 4; ++j) {
      async_ld_b128(a_dst + slot * (128 * 64) + j * 8, a_off0 + kb + j * 16u, A);
      async_ld_b128(w_dst + slot * (128 * 64) + j * 8, w_off0 + kb + j * 16u, W);
    }
  };

  v8f acc[2][4];
#pragma unroll
  for (int i = 0; i < 2; ++i)
#pragma unroll
    for (int j = 0; j < 4; ++j) acc[i][j] = v8f{};

  const int ns = K >> 6;               // 64-K stages
  issue(0, 0);
  if (ns > 1) issue(1, 64);

  for (int i = 0; i < ns; ++i) {
    if (i + 1 < ns) ASYNC_WAIT_8();    // our stage-i DMAs complete
    else            ASYNC_WAIT_0();
    __syncthreads();                   // whole block's stage i resident

    const int buf = i & 1;
    const __bf16* as = As[buf];
    const __bf16* bs = Bs[buf];
#pragma unroll
    for (int ks = 0; ks < 2; ++ks) {   // two K=32 substeps per stage
      v16bf bfr[4];
#pragma unroll
      for (int tj = 0; tj < 4; ++tj)
        bfr[tj] = *(const v16bf*)
            &bs[(wn + tj * 16 + lcol) * 64 + ks * 32 + lhalf * 16];
#pragma unroll
      for (int ti = 0; ti < 2; ++ti) {
        v16bf af = *(const v16bf*)
            &as[(wm + ti * 16 + lcol) * 64 + ks * 32 + lhalf * 16];
#pragma unroll
        for (int tj = 0; tj < 4; ++tj)
          acc[ti][tj] = wmma_bf16(af, bfr[tj], acc[ti][tj]);
      }
    }
    __syncthreads();                   // everyone done reading buf
    if (i + 2 < ns) issue(buf, (i + 2) * 64);
  }

  // branch-free epilogue (MODE is compile-time)
#pragma unroll
  for (int ti = 0; ti < 2; ++ti) {
#pragma unroll
    for (int tj = 0; tj < 4; ++tj) {
      const int row0 = m_blk + wm + ti * 16 + lhalf * 8;
      const int col  = n_blk + wn + tj * 16 + lcol;
#pragma unroll
      for (int r = 0; r < 8; ++r) {
        const size_t idx = (size_t)(row0 + r) * N + col;
        const float v = acc[ti][tj][r];
        if (MODE == 0) {
          C[idx] = v;
        } else if (MODE == 1) {
          C[idx] = R[idx] + v;
        } else {
          const float t = v > 0.f ? v : 0.f;
          Cbf[idx] = (__bf16)(t * t);
        }
      }
    }
  }
}

// ---------------------------------------------------------------------------
// f32 -> bf16 weight conversion (4 elements/thread; n % 1024 == 0)
// ---------------------------------------------------------------------------
__global__ __launch_bounds__(256)
void cvt_bf16_k(const float* __restrict__ src, __bf16* __restrict__ dst) {
  const size_t i = ((size_t)blockIdx.x * 256 + threadIdx.x) * 4;
  const float4 f = *(const float4*)(src + i);
  v4bf o;
  o[0] = (__bf16)f.x; o[1] = (__bf16)f.y; o[2] = (__bf16)f.z; o[3] = (__bf16)f.w;
  *(v4bf*)(dst + i) = o;
}

// ---------------------------------------------------------------------------
// Embedding gather: h[row,:] = embed[x[row],:]
// ---------------------------------------------------------------------------
__global__ __launch_bounds__(256)
void embed_k(const int* __restrict__ x, const float* __restrict__ E,
             float* __restrict__ H) {
  const int row = blockIdx.x;
  const int tok = x[row];
  const float2* src = (const float2*)(E + (size_t)tok * D_MODEL);
  float2* dst = (float2*)(H + (size_t)row * D_MODEL);
  dst[threadIdx.x] = src[threadIdx.x];
}

// ---------------------------------------------------------------------------
// Bigram gather: Abg[row,:] = bf16(bg_embed[prev(row) % NBIG, :])
// ---------------------------------------------------------------------------
__global__ __launch_bounds__(128)
void bg_gather_k(const int* __restrict__ x, const float* __restrict__ BG,
                 __bf16* __restrict__ A) {
  const int row = blockIdx.x;
  const int t = row & (SEQ - 1);
  int prev = (t == 0) ? 0 : x[row - 1];
  prev &= (NBIG - 1);
  A[(size_t)row * DBIG + threadIdx.x] =
      (__bf16)BG[(size_t)prev * DBIG + threadIdx.x];
}

// ---------------------------------------------------------------------------
// RMSNorm: y = bf16( x * rsqrt(mean(x^2)+eps) * w )   (one row per block)
// ---------------------------------------------------------------------------
__global__ __launch_bounds__(256)
void rmsnorm_k(const float* __restrict__ X, const float* __restrict__ w,
               __bf16* __restrict__ Y) {
  __shared__ float red[256];
  const int row = blockIdx.x;
  const int tid = threadIdx.x;
  const float* x = X + (size_t)row * D_MODEL;
  const float a0 = x[tid], a1 = x[tid + 256];
  red[tid] = a0 * a0 + a1 * a1;
  __syncthreads();
  for (int s = 128; s > 0; s >>= 1) {
    if (tid < s) red[tid] += red[tid + s];
    __syncthreads();
  }
  const float inv = rsqrtf(red[0] * (1.f / D_MODEL) + 1e-6f);
  Y[(size_t)row * D_MODEL + tid]       = (__bf16)(a0 * inv * w[tid]);
  Y[(size_t)row * D_MODEL + tid + 256] = (__bf16)(a1 * inv * w[tid + 256]);
}

// ---------------------------------------------------------------------------
// RoPE + head split.  qkv[row, 3*512] -> Q,K: [B,H,T,DH] bf16 (roped),
// V transposed: [B,H,DH,T] bf16.  Pairs (i,i+32), angle t*10000^(-2i/64).
// ---------------------------------------------------------------------------
__global__ __launch_bounds__(256)
void rope_split_k(const float* __restrict__ qkv, __bf16* __restrict__ Q,
                  __bf16* __restrict__ Kb, __bf16* __restrict__ Vt) {
  const int row = blockIdx.x;           // b*SEQ + t
  const int t = row & (SEQ - 1);
  const int b = row >> 10;
  const int h = threadIdx.x >> 5;       // 0..7
  const int i = threadIdx.x & 31;       // 0..31
  const float* base = qkv + (size_t)row * (3 * D_MODEL) + h * DHEAD;
  const float q1 = base[i],                 q2 = base[i + 32];
  const float k1 = base[D_MODEL + i],       k2 = base[D_MODEL + i + 32];
  const float v1 = base[2 * D_MODEL + i],   v2 = base[2 * D_MODEL + i + 32];
  const float freq = __expf(-0.28782158f * (float)i);  // 10000^(-i/32)
  const float ang = (float)t * freq;
  const float cv = cosf(ang), sv = sinf(ang);
  const size_t bh = (size_t)(b * NHEAD + h);
  const size_t qoff = (bh * SEQ + t) * DHEAD;
  Q[qoff + i]       = (__bf16)(q1 * cv - q2 * sv);
  Q[qoff + i + 32]  = (__bf16)(q1 * sv + q2 * cv);
  Kb[qoff + i]      = (__bf16)(k1 * cv - k2 * sv);
  Kb[qoff + i + 32] = (__bf16)(k1 * sv + k2 * cv);
  const size_t voff = bh * DHEAD * SEQ + t;
  Vt[voff + (size_t)i * SEQ]        = (__bf16)v1;
  Vt[voff + (size_t)(i + 32) * SEQ] = (__bf16)v2;
}

// ---------------------------------------------------------------------------
// Flash-style causal attention with self-key removal (WMMA for QK^T and PV).
// Grid: (B*H, SEQ/64), 128 thr = 4 waves, wave owns 16 query rows.
// ---------------------------------------------------------------------------
__global__ __launch_bounds__(128)
void attn_k(const __bf16* __restrict__ Q, const __bf16* __restrict__ Kb,
            const __bf16* __restrict__ Vt, __bf16* __restrict__ ctx) {
  __shared__ __bf16 Plds[4][16 * 32];

  const int bh = blockIdx.x;
  const int b = bh >> 3, h = bh & 7;
  const int qt = blockIdx.y;
  const int tid = threadIdx.x;
  const int wid = tid >> 5, lane = tid & 31;
  const int lcol = lane & 15, lhalf = lane >> 4;
  const int q0 = qt * 64 + wid * 16;

  const __bf16* qbase = Q  + (size_t)bh * SEQ * DHEAD;
  const __bf16* kbase = Kb + (size_t)bh * SEQ * DHEAD;
  const __bf16* vbase = Vt + (size_t)bh * DHEAD * SEQ;

  const int qr = q0 + lcol;
  const v16bf aq0 = *(const v16bf*)&qbase[(size_t)qr * DHEAD + lhalf * 16];
  const v16bf aq1 = *(const v16bf*)&qbase[(size_t)qr * DHEAD + 32 + lhalf * 16];

  v8f o[4];
#pragma unroll
  for (int nt = 0; nt < 4; ++nt) o[nt] = v8f{};
  float mrow[8], lrow[8];
#pragma unroll
  for (int r = 0; r < 8; ++r) { mrow[r] = -1e30f; lrow[r] = 0.f; }

  const float scale = 0.125f;           // DH^-0.5
  const int nchunk = (qt + 1) * 2;      // uniform per block

  for (int c = 0; c < nchunk; ++c) {
    const int kc = c * 32;
    v8f s[2];
#pragma unroll
    for (int kt = 0; kt < 2; ++kt) {
      const int kr = kc + kt * 16 + lcol;
      const v16bf bk0 = *(const v16bf*)&kbase[(size_t)kr * DHEAD + lhalf * 16];
      const v16bf bk1 = *(const v16bf*)&kbase[(size_t)kr * DHEAD + 32 + lhalf * 16];
      v8f sv = v8f{};
      sv = wmma_bf16(aq0, bk0, sv);
      sv = wmma_bf16(aq1, bk1, sv);
      s[kt] = sv;
    }

    float cm[8];
#pragma unroll
    for (int r = 0; r < 8; ++r) cm[r] = -1e30f;
#pragma unroll
    for (int kt = 0; kt < 2; ++kt) {
      const int key = kc + kt * 16 + lcol;
#pragma unroll
      for (int r = 0; r < 8; ++r) {
        const int rowi = q0 + lhalf * 8 + r;
        float v = s[kt][r] * scale;
        if (key > rowi) v = -1e30f;
        s[kt][r] = v;
        cm[r] = fmaxf(cm[r], v);
      }
    }
#pragma unroll
    for (int r = 0; r < 8; ++r) {
      float v = cm[r];
      v = fmaxf(v, __shfl_xor(v, 1, 16));
      v = fmaxf(v, __shfl_xor(v, 2, 16));
      v = fmaxf(v, __shfl_xor(v, 4, 16));
      v = fmaxf(v, __shfl_xor(v, 8, 16));
      cm[r] = v;
    }

#pragma unroll
    for (int r = 0; r < 8; ++r) {
      const float mnew  = fmaxf(mrow[r], cm[r]);
      const float alpha = __expf(mrow[r] - mnew);
      mrow[r] = mnew;
      float lsum = 0.f;
#pragma unroll
      for (int kt = 0; kt < 2; ++kt) {
        const float pv = __expf(s[kt][r] - mnew);
        s[kt][r] = pv;
        lsum += pv;
      }
      lsum += __shfl_xor(lsum, 1, 16);
      lsum += __shfl_xor(lsum, 2, 16);
      lsum += __shfl_xor(lsum, 4, 16);
      lsum += __shfl_xor(lsum, 8, 16);
      lrow[r] = lrow[r] * alpha + lsum;
#pragma unroll
      for (int nt = 0; nt < 4; ++nt) o[nt][r] *= alpha;
    }

    __syncthreads();
#pragma unroll
    for (int kt = 0; kt < 2; ++kt) {
      const int key = kc + kt * 16 + lcol;
#pragma unroll
      for (int r = 0; r < 8; ++r) {
        const int rowi = q0 + lhalf * 8 + r;
        const float pv = (key == rowi) ? 0.f : s[kt][r];
        Plds[wid][(lhalf * 8 + r) * 32 + kt * 16 + lcol] = (__bf16)pv;
      }
    }
    __syncthreads();
    const v16bf pa = *(const v16bf*)&Plds[wid][lcol * 32 + lhalf * 16];

#pragma unroll
    for (int nt = 0; nt < 4; ++nt) {
      const v16bf bv = *(const v16bf*)
          &vbase[(size_t)(nt * 16 + lcol) * SEQ + kc + lhalf * 16];
      o[nt] = wmma_bf16(pa, bv, o[nt]);
    }
  }

#pragma unroll
  for (int nt = 0; nt < 4; ++nt) {
    const int dh = nt * 16 + lcol;
#pragma unroll
    for (int r = 0; r < 8; ++r) {
      const int t = q0 + lhalf * 8 + r;
      const float v = o[nt][r] / lrow[r];
      ctx[(((size_t)b * SEQ + t) * NHEAD + h) * DHEAD + dh] = (__bf16)v;
    }
  }
}

// ---------------------------------------------------------------------------
// Host-side orchestration (graph-capture safe: only launches on stream)
// ---------------------------------------------------------------------------
extern "C" void kernel_launch(void* const* d_in, const int* in_sizes, int n_in,
                              void* d_out, int out_size, void* d_ws,
                              size_t ws_size, hipStream_t stream) {
  (void)in_sizes; (void)n_in; (void)out_size; (void)ws_size;
  const int*   x     = (const int*)  d_in[0];
  const float* embed = (const float*)d_in[1];
  const float* Wqkv  = (const float*)d_in[2];
  const float* Wproj = (const float*)d_in[3];
  const float* n1w   = (const float*)d_in[4];
  const float* n2w   = (const float*)d_in[5];
  const float* Wfc1  = (const float*)d_in[6];
  const float* Wfc2  = (const float*)d_in[7];
  const float* nfw   = (const float*)d_in[8];
  const float* bgE   = (const float*)d_in[9];
  const float* bgP   = (const float*)d_in[10];
  float* out = (float*)d_out;

  char* ws = (char*)d_ws;
  size_t off = 0;
  auto carve = [&](size_t bytes) {
    void* p = ws + off;
    off += (bytes + 255) & ~(size_t)255;
    return p;
  };
  float*  h      = (float*)  carve((size_t)ROWS * D_MODEL * 4);
  float*  qkv    = (float*)  carve((size_t)ROWS * 3 * D_MODEL * 4);
  float*  bigram = (float*)  carve((size_t)ROWS * VOCAB * 4);
  __bf16* xn     = (__bf16*) carve((size_t)ROWS * D_MODEL * 2);
  __bf16* mlp    = (__bf16*) carve((size_t)ROWS * HIDDEN * 2);
  __bf16* qb     = (__bf16*) carve((size_t)ROWS * D_MODEL * 2);
  __bf16* kb     = (__bf16*) carve((size_t)ROWS * D_MODEL * 2);
  __bf16* vt     = (__bf16*) carve((size_t)ROWS * D_MODEL * 2);
  __bf16* ctx    = (__bf16*) carve((size_t)ROWS * D_MODEL * 2);
  __bf16* bga    = (__bf16*) carve((size_t)ROWS * DBIG * 2);
  const size_t nQKV = (size_t)NLAYER * 3 * D_MODEL * D_MODEL;
  const size_t nPRJ = (size_t)NLAYER * D_MODEL * D_MODEL;
  const size_t nFC  = (size_t)NLAYER * HIDDEN * D_MODEL;
  __bf16* wqkv_bf  = (__bf16*) carve(nQKV * 2);
  __bf16* wproj_bf = (__bf16*) carve(nPRJ * 2);
  __bf16* wfc1_bf  = (__bf16*) carve(nFC * 2);
  __bf16* wfc2_bf  = (__bf16*) carve(nFC * 2);
  __bf16* embed_bf = (__bf16*) carve((size_t)VOCAB * D_MODEL * 2);
  __bf16* bgp_bf   = (__bf16*) carve((size_t)VOCAB * DBIG * 2);

  cvt_bf16_k<<<nQKV / 1024, 256, 0, stream>>>(Wqkv, wqkv_bf);
  cvt_bf16_k<<<nPRJ / 1024, 256, 0, stream>>>(Wproj, wproj_bf);
  cvt_bf16_k<<<nFC  / 1024, 256, 0, stream>>>(Wfc1, wfc1_bf);
  cvt_bf16_k<<<nFC  / 1024, 256, 0, stream>>>(Wfc2, wfc2_bf);
  cvt_bf16_k<<<(size_t)VOCAB * D_MODEL / 1024, 256, 0, stream>>>(embed, embed_bf);
  cvt_bf16_k<<<(size_t)VOCAB * DBIG / 1024, 256, 0, stream>>>(bgP, bgp_bf);

  embed_k<<<ROWS, 256, 0, stream>>>(x, embed, h);
  bg_gather_k<<<ROWS, DBIG, 0, stream>>>(x, bgE, bga);
  gemm_bf16_k<0><<<dim3(VOCAB / 128, ROWS / 128), 256, 0, stream>>>(
      bga, bgp_bf, bigram, nullptr, nullptr, VOCAB, DBIG);

  for (int l = 0; l < NLAYER; ++l) {
    rmsnorm_k<<<ROWS, 256, 0, stream>>>(h, n1w + l * D_MODEL, xn);
    gemm_bf16_k<0><<<dim3(3 * D_MODEL / 128, ROWS / 128), 256, 0, stream>>>(
        xn, wqkv_bf + (size_t)l * 3 * D_MODEL * D_MODEL, qkv, nullptr, nullptr,
        3 * D_MODEL, D_MODEL);
    rope_split_k<<<ROWS, 256, 0, stream>>>(qkv, qb, kb, vt);
    attn_k<<<dim3(BATCH * NHEAD, SEQ / 64), 128, 0, stream>>>(qb, kb, vt, ctx);
    gemm_bf16_k<1><<<dim3(D_MODEL / 128, ROWS / 128), 256, 0, stream>>>(
        ctx, wproj_bf + (size_t)l * D_MODEL * D_MODEL, h, h, nullptr,
        D_MODEL, D_MODEL);
    rmsnorm_k<<<ROWS, 256, 0, stream>>>(h, n2w + l * D_MODEL, xn);
    gemm_bf16_k<2><<<dim3(HIDDEN / 128, ROWS / 128), 256, 0, stream>>>(
        xn, wfc1_bf + (size_t)l * HIDDEN * D_MODEL, nullptr, nullptr, mlp,
        HIDDEN, D_MODEL);
    gemm_bf16_k<1><<<dim3(D_MODEL / 128, ROWS / 128), 256, 0, stream>>>(
        mlp, wfc2_bf + (size_t)l * D_MODEL * HIDDEN, h, h, nullptr,
        D_MODEL, HIDDEN);
  }

  rmsnorm_k<<<ROWS, 256, 0, stream>>>(h, nfw, xn);
  gemm_bf16_k<1><<<dim3(VOCAB / 128, ROWS / 128), 256, 0, stream>>>(
      xn, embed_bf, out, bigram, nullptr, VOCAB, D_MODEL);
}
